// MGEL_56547539419680
// MI455X (gfx1250) — compile-verified, matrix operand order checked by reading, not run on hardware
//
#include <hip/hip_runtime.h>
#include <hip/hip_bf16.h>

typedef __attribute__((ext_vector_type(16))) __bf16 v16bf;
typedef __attribute__((ext_vector_type(8)))  float  v8f;
typedef __attribute__((ext_vector_type(4)))  unsigned int v4u;

#define TPB 256
#define SB_LD 40   // LDS row pitch in ushorts (80B) -> conflict-free 16-row reads

struct Q2 { v4u lo, hi; };   // 32B -> one v16bf fragment

__device__ __forceinline__ unsigned short f2bf(float f) {
  unsigned u = __float_as_uint(f);
  u += 0x7FFFu + ((u >> 16) & 1u);           // round-to-nearest-even
  return (unsigned short)(u >> 16);
}
__device__ __forceinline__ float bf2f(unsigned short h) {
  return __uint_as_float(((unsigned)h) << 16);
}
__device__ __forceinline__ v16bf ldfrag(const unsigned short* p) {
  Q2 q;
  q.lo = *(const v4u*)(p);
  q.hi = *(const v4u*)(p + 16);   // k = 16..23 lives 16 ushorts later
  return __builtin_bit_cast(v16bf, q);
}

// ---------------------------------------------------------------------------
// WMMA GEMM: C[M,N] = A[M,K] * B[K,N] (+bias)(+relu)
//   A  : bf16, row-major, leading dim lda (mult of 64, zero-padded past K)
//   Bt : bf16, TRANSPOSED weights [Np x ldb], ldb mult of 64, zero-padded
//   block = 256 threads = 8 waves -> 256x64 output tile, all waves share one
//   64-col strip; B tile (64x32) staged in LDS once per k-step (double-buffered),
//   each wave computes 32x64 with 8 wmma per k-step; A streams from global.
// ---------------------------------------------------------------------------
__global__ void __launch_bounds__(TPB) k_gemm_bf16(
    const unsigned short* __restrict__ A, int lda,
    const unsigned short* __restrict__ Bt, int ldb,
    const float* __restrict__ bias, float* __restrict__ Cf, int ldcf,
    unsigned short* __restrict__ Cbf, int ldcbf, int M, int N, int K, int relu) {
  __shared__ unsigned short sB[2][64 * SB_LD];
  const int tilesN4 = (N + 63) >> 6;
  const int bm = blockIdx.x / tilesN4;          // 256-row block index
  const int tn = blockIdx.x % tilesN4;
  const int n0 = tn << 6;
  const int tid = threadIdx.x;
  const int wv = tid >> 5, lane = tid & 31;
  const int hh = lane >> 4, l15 = lane & 15;
  const int m0 = bm * 256 + wv * 32;
  int am0 = m0 + l15;      if (am0 >= M) am0 = M - 1;   // M mult of 256 in practice
  int am1 = m0 + 16 + l15; if (am1 >= M) am1 = M - 1;
  const unsigned short* __restrict__ ar0 = A + (size_t)am0 * lda + (hh << 3);
  const unsigned short* __restrict__ ar1 = A + (size_t)am1 * lda + (hh << 3);
  // cooperative B staging: each thread moves one b128 (64 rows x 32 ks = 256 o16x8)
  const int srow = tid >> 2;                    // 0..63
  const int skseg = (tid & 3) << 3;             // 0,8,16,24
  const unsigned short* __restrict__ bsrc = Bt + (size_t)(n0 + srow) * ldb + skseg;
  unsigned short* const sdst0 = &sB[0][srow * SB_LD + skseg];
  unsigned short* const sdst1 = &sB[1][srow * SB_LD + skseg];
  const int Kc = (K + 31) & ~31;
  const int steps = Kc >> 5;
  const v8f vz = {0.f, 0.f, 0.f, 0.f, 0.f, 0.f, 0.f, 0.f};
  v8f acc[8] = {vz, vz, vz, vz, vz, vz, vz, vz};      // [s*4 + t]

  // prologue: stage k-block 0 into LDS buffer 0; prime A registers
  *(v4u*)sdst0 = *(const v4u*)(bsrc);
  v16bf a0c = ldfrag(ar0);
  v16bf a1c = ldfrag(ar1);
  __syncthreads();

  for (int i = 0; i < steps; ++i) {
    const int p = i & 1;
    const bool more = (i + 1) < steps;          // uniform across block
    v4u bnext = {0u, 0u, 0u, 0u};
    v16bf a0n = a0c, a1n = a1c;
    if (more) {                                 // issue next-block global loads early
      const int ko = (i + 1) << 5;
      bnext = *(const v4u*)(bsrc + ko);
      a0n = ldfrag(ar0 + ko);
      a1n = ldfrag(ar1 + ko);
    }
    // B fragments from LDS (rows t*16+l15, pitch SB_LD; conflict-free)
    const unsigned short* sbase = &sB[p][l15 * SB_LD + (hh << 3)];
    const v16bf b0 = ldfrag(sbase);
    const v16bf b1 = ldfrag(sbase + 16 * SB_LD);
    const v16bf b2 = ldfrag(sbase + 32 * SB_LD);
    const v16bf b3 = ldfrag(sbase + 48 * SB_LD);
    acc[0] = __builtin_amdgcn_wmma_f32_16x16x32_bf16(false, a0c, false, b0, (short)0, acc[0], false, false);
    acc[1] = __builtin_amdgcn_wmma_f32_16x16x32_bf16(false, a0c, false, b1, (short)0, acc[1], false, false);
    acc[2] = __builtin_amdgcn_wmma_f32_16x16x32_bf16(false, a0c, false, b2, (short)0, acc[2], false, false);
    acc[3] = __builtin_amdgcn_wmma_f32_16x16x32_bf16(false, a0c, false, b3, (short)0, acc[3], false, false);
    acc[4] = __builtin_amdgcn_wmma_f32_16x16x32_bf16(false, a1c, false, b0, (short)0, acc[4], false, false);
    acc[5] = __builtin_amdgcn_wmma_f32_16x16x32_bf16(false, a1c, false, b1, (short)0, acc[5], false, false);
    acc[6] = __builtin_amdgcn_wmma_f32_16x16x32_bf16(false, a1c, false, b2, (short)0, acc[6], false, false);
    acc[7] = __builtin_amdgcn_wmma_f32_16x16x32_bf16(false, a1c, false, b3, (short)0, acc[7], false, false);
    if (more) {
      *(v4u*)(p ? sdst0 : sdst1) = bnext;       // fill other buffer
      a0c = a0n; a1c = a1n;
    }
    __syncthreads();                            // uniform barrier each iteration
  }

#pragma unroll
  for (int s = 0; s < 2; ++s) {
#pragma unroll
    for (int t = 0; t < 4; ++t) {
      const int n = n0 + (t << 4) + l15;
      const bool nin = (n < N);
      const float bv = (nin && bias) ? bias[n] : 0.0f;
#pragma unroll
      for (int r = 0; r < 8; ++r) {
        const int m = m0 + (s << 4) + (hh << 3) + r;  // C/D layout: vgpr r -> M = r + 8*half
        if (m >= M) continue;
        float v = 0.0f;
        if (nin) {
          v = acc[s * 4 + t][r] + bv;
          if (relu) v = fmaxf(v, 0.0f);
          if (Cf) Cf[(size_t)m * ldcf + n] = v;
        }
        if (Cbf) Cbf[(size_t)m * ldcbf + n] = f2bf(v);   // zero-fills padding cols
      }
    }
  }
}

// ---------------------------------------------------------------------------
// converters (write zero-padded bf16 with leading dim ldd)
// ---------------------------------------------------------------------------
__global__ void k_cvt_pad(const float* __restrict__ s, unsigned short* __restrict__ d,
                          long long rows, int cols, int ldd) {
  for (long long t = (long long)blockIdx.x * blockDim.x + threadIdx.x; t < rows * ldd;
       t += (long long)gridDim.x * blockDim.x) {
    long long r = t / ldd; int c = (int)(t % ldd);
    d[t] = (c < cols) ? f2bf(s[(size_t)r * cols + c]) : (unsigned short)0;
  }
}
// weights: w[K,N] f32 -> Bt[Np, Kp] bf16 (transposed, zero-padded)
__global__ void k_wcvt(const float* __restrict__ w, unsigned short* __restrict__ bt,
                       int K, int N, int Kp, long long total) {
  for (long long t = (long long)blockIdx.x * blockDim.x + threadIdx.x; t < total;
       t += (long long)gridDim.x * blockDim.x) {
    int n = (int)(t / Kp), k = (int)(t % Kp);
    bt[t] = (n < N && k < K) ? f2bf(w[(size_t)k * N + n]) : (unsigned short)0;
  }
}
__global__ void k_gather_cvt(const float* __restrict__ src, const int* __restrict__ idx,
                             unsigned short* __restrict__ dst, long long rows, int cols, int ldd) {
  for (long long t = (long long)blockIdx.x * blockDim.x + threadIdx.x; t < rows * ldd;
       t += (long long)gridDim.x * blockDim.x) {
    long long r = t / ldd; int c = (int)(t % ldd);
    dst[t] = (c < cols) ? f2bf(src[(size_t)idx[r] * cols + c]) : (unsigned short)0;
  }
}
__global__ void k_gather2_cvt(const float* __restrict__ s1, int c1,
                              const float* __restrict__ s2, int c2,
                              const int* __restrict__ idx,
                              unsigned short* __restrict__ dst, long long rows, int ldd) {
  int ct = c1 + c2;
  for (long long t = (long long)blockIdx.x * blockDim.x + threadIdx.x; t < rows * ldd;
       t += (long long)gridDim.x * blockDim.x) {
    long long r = t / ldd; int c = (int)(t % ldd);
    float v = 0.f;
    if (c < ct) {
      int g = idx[r];
      v = (c < c1) ? s1[(size_t)g * c1 + c] : s2[(size_t)g * c2 + (c - c1)];
    }
    dst[t] = (c < ct) ? f2bf(v) : (unsigned short)0;
  }
}
// dst[n, ldd] = pad([h+y, h-y, h]) with y = supp[batch[n]]
__global__ void k_suppcat(const float* __restrict__ h, const float* __restrict__ supp,
                          const int* __restrict__ batch, unsigned short* __restrict__ dst,
                          long long n, int F, int ldd) {
  int F3 = 3 * F;
  for (long long t = (long long)blockIdx.x * blockDim.x + threadIdx.x; t < n * ldd;
       t += (long long)gridDim.x * blockDim.x) {
    long long node = t / ldd; int f = (int)(t % ldd);
    if (f >= F3) { dst[t] = 0; continue; }
    int ff = f % F;
    float hv = h[(size_t)node * F + ff];
    float yv = supp[(size_t)batch[node] * F + ff];
    float v = (f < F) ? (hv + yv) : (f < 2 * F ? (hv - yv) : hv);
    dst[t] = f2bf(v);
  }
}
__global__ void k_catbf(const unsigned short* __restrict__ a, int ca,
                        const unsigned short* __restrict__ b, int cb,
                        unsigned short* __restrict__ dst, long long rows, int ldd) {
  int ct = ca + cb;
  for (long long t = (long long)blockIdx.x * blockDim.x + threadIdx.x; t < rows * ldd;
       t += (long long)gridDim.x * blockDim.x) {
    long long r = t / ldd; int c = (int)(t % ldd);
    unsigned short v = 0;
    if (c < ca) v = a[(size_t)r * ca + c];
    else if (c < ct) v = b[(size_t)r * cb + (c - ca)];
    dst[t] = v;
  }
}

// ---------------------------------------------------------------------------
// graph / elementwise kernels (fp32, grid-stride)
// ---------------------------------------------------------------------------
__global__ void k_fill(float* p, float v, long long n) {
  for (long long i = (long long)blockIdx.x * blockDim.x + threadIdx.x; i < n;
       i += (long long)gridDim.x * blockDim.x) p[i] = v;
}
__global__ void k_deg(const int* __restrict__ col, const float* __restrict__ w,
                      float* __restrict__ deg, long long E) {
  for (long long e = (long long)blockIdx.x * blockDim.x + threadIdx.x; e < E;
       e += (long long)gridDim.x * blockDim.x)
    atomicAdd(&deg[col[e]], w ? w[e] : 1.0f);
}
__global__ void k_dinv(float* d, long long n) {
  for (long long i = (long long)blockIdx.x * blockDim.x + threadIdx.x; i < n;
       i += (long long)gridDim.x * blockDim.x) {
    float v = d[i];
    d[i] = (v > 0.f) ? rsqrtf(fmaxf(v, 1e-12f)) : 0.f;
  }
}
__global__ void k_outinit(const float* __restrict__ h, const float* __restrict__ dinv,
                          const float* __restrict__ bias, float* __restrict__ out,
                          long long n, int F) {
  for (long long t = (long long)blockIdx.x * blockDim.x + threadIdx.x; t < n * F;
       t += (long long)gridDim.x * blockDim.x) {
    long long i = t / F; int f = (int)(t % F);
    float di = dinv[i];
    out[t] = h[t] * di * di + bias[f];
  }
}
__global__ void k_scatter(const int* __restrict__ row, const int* __restrict__ col,
                          const float* __restrict__ w, const float* __restrict__ dinv,
                          const float* __restrict__ h, float* __restrict__ out,
                          long long E, int F) {
  for (long long t = (long long)blockIdx.x * blockDim.x + threadIdx.x; t < E * F;
       t += (long long)gridDim.x * blockDim.x) {
    long long e = t / F; int f = (int)(t % F);
    int r = row[e], c = col[e];
    float ww = w ? w[e] : 1.0f;
    atomicAdd(&out[(size_t)c * F + f], h[(size_t)r * F + f] * dinv[r] * ww * dinv[c]);
  }
}
// relu in place (x: rows x cols contiguous); optional padded bf16 emit
__global__ void k_relu(float* __restrict__ x, unsigned short* __restrict__ xbf,
                       long long rows, int cols, int ldd) {
  for (long long t = (long long)blockIdx.x * blockDim.x + threadIdx.x; t < rows * ldd;
       t += (long long)gridDim.x * blockDim.x) {
    long long r = t / ldd; int c = (int)(t % ldd);
    if (c < cols) {
      size_t ix = (size_t)r * cols + c;
      float v = fmaxf(x[ix], 0.0f);
      x[ix] = v;
      if (xbf) xbf[t] = f2bf(v);
    } else if (xbf) xbf[t] = 0;
  }
}
__global__ void k_colmean(const float* __restrict__ x, float* __restrict__ mean,
                          int rows, int cols) {
  int c = blockIdx.x * blockDim.x + threadIdx.x;
  if (c >= cols) return;
  float s = 0.f;
  for (int r = 0; r < rows; ++r) s += x[(size_t)r * cols + c];
  mean[c] = s / rows;
}
__global__ void k_ssq(const float* __restrict__ x, const float* __restrict__ mean,
                      float* __restrict__ ssq, long long rows, int cols) {
  float s = 0.f;
  for (long long t = (long long)blockIdx.x * blockDim.x + threadIdx.x; t < rows * cols;
       t += (long long)gridDim.x * blockDim.x) {
    int c = (int)(t % cols);
    float v = x[t] - mean[c];
    s += v * v;
  }
  atomicAdd(ssq, s);
}
// y = relu((x - colmean) * rsqrt(1e-5 + ssq/rows)); yf exact, ybf padded(ldd)
__global__ void k_pnorm(const float* __restrict__ x, const float* __restrict__ mean,
                        const float* __restrict__ ssq, float* __restrict__ yf,
                        unsigned short* __restrict__ ybf, long long rows, int cols, int ldd) {
  float scale = rsqrtf(1e-5f + ssq[0] / (float)rows);
  for (long long t = (long long)blockIdx.x * blockDim.x + threadIdx.x; t < rows * ldd;
       t += (long long)gridDim.x * blockDim.x) {
    long long r = t / ldd; int c = (int)(t % ldd);
    if (c < cols) {
      float v = fmaxf((x[(size_t)r * cols + c] - mean[c]) * scale, 0.0f);
      if (yf)  yf[(size_t)r * cols + c] = v;
      if (ybf) ybf[t] = f2bf(v);
    } else if (ybf) ybf[t] = 0;
  }
}
__global__ void k_count(const int* __restrict__ batch, float* __restrict__ cnt, long long n) {
  for (long long i = (long long)blockIdx.x * blockDim.x + threadIdx.x; i < n;
       i += (long long)gridDim.x * blockDim.x) atomicAdd(&cnt[batch[i]], 1.0f);
}
__global__ void k_pooladd(const float* __restrict__ x, const int* __restrict__ batch,
                          float* __restrict__ pool, long long n, int F) {
  for (long long t = (long long)blockIdx.x * blockDim.x + threadIdx.x; t < n * F;
       t += (long long)gridDim.x * blockDim.x) {
    long long node = t / F; int f = (int)(t % F);
    atomicAdd(&pool[(size_t)batch[node] * F + f], x[t]);
  }
}
__global__ void k_pooldiv(float* __restrict__ pool, const float* __restrict__ cnt,
                          long long G, int F) {
  for (long long t = (long long)blockIdx.x * blockDim.x + threadIdx.x; t < G * F;
       t += (long long)gridDim.x * blockDim.x)
    pool[t] /= fmaxf(cnt[t / F], 1.0f);
}
// out[r*ldo + co] = sum_k bf(A[r,k]) * w[k] + b[0]
__global__ void k_dotout(const unsigned short* __restrict__ A, int lda,
                         const float* __restrict__ w, const float* __restrict__ b,
                         float* __restrict__ out, long long rows, int K, int ldo, int co) {
  for (long long r = (long long)blockIdx.x * blockDim.x + threadIdx.x; r < rows;
       r += (long long)gridDim.x * blockDim.x) {
    float acc = 0.f;
    const unsigned short* ap = A + (size_t)r * lda;
    for (int k = 0; k < K; ++k) acc += bf2f(ap[k]) * w[k];
    out[(size_t)r * ldo + co] = acc + b[0];
  }
}
__global__ void k_final9(const float* __restrict__ ans, const float* __restrict__ w,
                         const float* __restrict__ b, float* __restrict__ out, long long n) {
  for (long long r = (long long)blockIdx.x * blockDim.x + threadIdx.x; r < n;
       r += (long long)gridDim.x * blockDim.x) {
    float acc = 0.f;
#pragma unroll
    for (int j = 0; j < 9; ++j) acc += ans[r * 9 + j] * w[j];
    out[r] = acc + b[0];
  }
}

static inline int nblk(long long n) {
  long long b = (n + TPB - 1) / TPB;
  if (b < 1) b = 1;
  if (b > 1048576) b = 1048576;
  return (int)b;
}
static inline int P64i(int x) { return (x + 63) & ~63; }

extern "C" void kernel_launch(void* const* d_in, const int* in_sizes, int n_in,
                              void* d_out, int out_size, void* d_ws, size_t ws_size,
                              hipStream_t stream) {
  (void)in_sizes; (void)n_in; (void)out_size; (void)ws_size;
  const int N_DRUGS = 2048, N_TARGETS = 1024, N_PAIRS = 32768;
  const int ND = 65536, NT = 131072;
  const long long ED_MOL = 262144, ET_MOL = 524288, ED_SIM = 131072, ET_SIM = 65536;

  auto F32 = [&](int k) { return (const float*)d_in[k]; };
  auto I32 = [&](int k) { return (const int*)d_in[k]; };

  const float* drug_x   = F32(0);  const int* dsim_ei = I32(1);  const float* dsim_w = F32(2);
  const float* target_x = F32(3);  const int* tsim_ei = I32(4);  const float* tsim_w = F32(5);
  const float* mg_x     = F32(6);  const int* mg_ei   = I32(7);  const int* mg_batch = I32(8);
  const float* pg_x     = F32(9);  const int* pg_ei   = I32(10); const int* pg_batch = I32(11);
  const int* drug_id    = I32(12); const int* target_id = I32(13);
  const int p_dsim = 14, p_tsim = 20, p_dtr = 26, p_ttr = 30, p_dgcn = 34, p_tgcn = 40;
  const int p_ds = 46, p_dg = 50, p_da = 54, p_ts = 58, p_tg = 62, p_ta = 66;
  const int p_j = 70, p_mlp = 124;

  // ---- workspace arena ----
  char* ws = (char*)d_ws;
  size_t off = 0;
  auto AL = [&](size_t bytes) -> void* {
    off = (off + 255) & ~(size_t)255;
    void* p = ws + off; off += bytes; return p;
  };
  unsigned short* wbf  = (unsigned short*)AL((size_t)1024 * 1024 * 2);   // Np*Kp <= 1M elems
  float* FA = (float*)AL((size_t)NT * 216 * 4);
  float* FB = (float*)AL((size_t)NT * 216 * 4);
  float* FC = (float*)AL((size_t)NT * 216 * 4);
  unsigned short* ABF  = (unsigned short*)AL((size_t)N_PAIRS * 1024 * 2);
  unsigned short* ABF2 = (unsigned short*)AL((size_t)N_PAIRS * 1024 * 2);
  unsigned short* ABF3 = (unsigned short*)AL((size_t)N_PAIRS * 512 * 2);
  float* drug3   = (float*)AL((size_t)N_DRUGS * 256 * 4);
  float* target3 = (float*)AL((size_t)N_TARGETS * 256 * 4);
  float* drugS   = (float*)AL((size_t)N_DRUGS * 78 * 4);
  float* targetS = (float*)AL((size_t)N_TARGETS * 54 * 4);
  float* drugG   = (float*)AL((size_t)N_DRUGS * 312 * 4);
  float* targetG = (float*)AL((size_t)N_TARGETS * 216 * 4);
  unsigned short* dpair[3], *tpair[3];
  for (int i = 0; i < 3; ++i) dpair[i] = (unsigned short*)AL((size_t)N_PAIRS * 128 * 2);
  for (int i = 0; i < 3; ++i) tpair[i] = (unsigned short*)AL((size_t)N_PAIRS * 128 * 2);
  float* ansb = (float*)AL((size_t)N_PAIRS * 9 * 4);
  float* dinv = (float*)AL((size_t)NT * 4);
  float* cnt  = (float*)AL((size_t)N_DRUGS * 4);
  float* mean = (float*)AL((size_t)512 * 4);
  float* ssq  = (float*)AL(64);

  // gemm: A(bf16, lda=P64(K)) x W(d_in[widx], K x N) -> Cf(f32, ldcf) / Cbf(bf16, ldcbf=P64(N))
  auto gemm = [&](const unsigned short* Abf, int lda, int widx, int M, int N, int K,
                  bool useBias, bool relu, float* Cf, int ldcf,
                  unsigned short* Cbf, int ldcbf) {
    int Kp = P64i(K), Np = P64i(N);
    k_wcvt<<<nblk((long long)Np * Kp), TPB, 0, stream>>>(F32(widx), wbf, K, N, Kp,
                                                         (long long)Np * Kp);
    const float* bias = useBias ? F32(widx + 1) : nullptr;
    int blocks = ((M + 255) / 256) * (Np / 64);
    k_gemm_bf16<<<blocks, TPB, 0, stream>>>(Abf, lda, wbf, Kp, bias, Cf, ldcf, Cbf, ldcbf,
                                            M, N, K, relu ? 1 : 0);
  };

  // ---- similarity GCN chains (GCNConv -> PairNorm -> ReLU) x3 ----
  auto simStage = [&](const float* x0, int n, int din, const int* ei, const float* ew,
                      long long E, int pbase, float* out3) {
    k_fill<<<nblk(n), TPB, 0, stream>>>(dinv, 1.0f, n);                 // self loops
    k_deg<<<nblk(E), TPB, 0, stream>>>(ei + E, ew, dinv, E);
    k_dinv<<<nblk(n), TPB, 0, stream>>>(dinv, n);
    int dims[4] = {din, 512, 512, 256};
    k_cvt_pad<<<nblk((long long)n * P64i(din)), TPB, 0, stream>>>(x0, ABF, n, din, P64i(din));
    for (int l = 0; l < 3; ++l) {
      int K = dims[l], N = dims[l + 1];
      gemm(ABF, P64i(K), pbase + 2 * l, n, N, K, false, false, FB, N, nullptr, 0);
      k_outinit<<<nblk((long long)n * N), TPB, 0, stream>>>(FB, dinv, F32(pbase + 2 * l + 1),
                                                            FC, n, N);
      k_scatter<<<nblk(E * N), TPB, 0, stream>>>(ei, ei + E, ew, dinv, FB, FC, E, N);
      hipMemsetAsync(ssq, 0, 4, stream);
      k_colmean<<<nblk(N), TPB, 0, stream>>>(FC, mean, n, N);
      k_ssq<<<nblk((long long)n * N), TPB, 0, stream>>>(FC, mean, ssq, n, N);
      float* yf = (l == 2) ? out3 : nullptr;
      unsigned short* ybf = (l == 2) ? nullptr : ABF;
      k_pnorm<<<nblk((long long)n * P64i(N)), TPB, 0, stream>>>(FC, mean, ssq, yf, ybf, n, N,
                                                                P64i(N));
    }
  };
  simStage(drug_x, N_DRUGS, 384, dsim_ei, dsim_w, ED_SIM, p_dsim, drug3);
  simStage(target_x, N_TARGETS, 768, tsim_ei, tsim_w, ET_SIM, p_tsim, target3);

  // ---- transform MLPs (supplements) ----
  k_cvt_pad<<<nblk((long long)N_DRUGS * 256), TPB, 0, stream>>>(drug3, ABF, N_DRUGS, 256, 256);
  gemm(ABF, 256, p_dtr, N_DRUGS, 1024, 256, true, true, nullptr, 0, ABF2, 1024);
  gemm(ABF2, 1024, p_dtr + 2, N_DRUGS, 78, 1024, true, false, drugS, 78, nullptr, 0);
  k_cvt_pad<<<nblk((long long)N_TARGETS * 256), TPB, 0, stream>>>(target3, ABF, N_TARGETS, 256, 256);
  gemm(ABF, 256, p_ttr, N_TARGETS, 1024, 256, true, true, nullptr, 0, ABF2, 1024);
  gemm(ABF2, 1024, p_ttr + 2, N_TARGETS, 54, 1024, true, false, targetS, 54, nullptr, 0);

  // ---- molecular graph GCN blocks ----
  auto molStage = [&](const float* x0, int n, int G, const int* ei, long long E,
                      const int* batch, int pbase, const float* supp, int F0, int F2out,
                      float* poolOut) {
    int F1 = 3 * F0;
    k_fill<<<nblk(n), TPB, 0, stream>>>(dinv, 1.0f, n);
    k_deg<<<nblk(E), TPB, 0, stream>>>(ei + E, nullptr, dinv, E);
    k_dinv<<<nblk(n), TPB, 0, stream>>>(dinv, n);
    // layer 0: F0 -> F0
    k_cvt_pad<<<nblk((long long)n * P64i(F0)), TPB, 0, stream>>>(x0, ABF, n, F0, P64i(F0));
    gemm(ABF, P64i(F0), pbase, n, F0, F0, false, false, FB, F0, nullptr, 0);
    k_outinit<<<nblk((long long)n * F0), TPB, 0, stream>>>(FB, dinv, F32(pbase + 1), FC, n, F0);
    k_scatter<<<nblk(E * F0), TPB, 0, stream>>>(ei, ei + E, nullptr, dinv, FB, FC, E, F0);
    k_relu<<<nblk((long long)n * F0), TPB, 0, stream>>>(FC, nullptr, n, F0, F0);
    // layer 1: cat(out+y,out-y,out) -> F1 -> F1
    k_suppcat<<<nblk((long long)n * P64i(F1)), TPB, 0, stream>>>(FC, supp, batch, ABF, n, F0,
                                                                 P64i(F1));
    gemm(ABF, P64i(F1), pbase + 2, n, F1, F1, false, false, FB, F1, nullptr, 0);
    k_outinit<<<nblk((long long)n * F1), TPB, 0, stream>>>(FB, dinv, F32(pbase + 3), FA, n, F1);
    k_scatter<<<nblk(E * F1), TPB, 0, stream>>>(ei, ei + E, nullptr, dinv, FB, FA, E, F1);
    k_relu<<<nblk((long long)n * P64i(F1)), TPB, 0, stream>>>(FA, ABF2, n, F1, P64i(F1));
    // layer 2: F1 -> F2out
    gemm(ABF2, P64i(F1), pbase + 4, n, F2out, F1, false, false, FB, F2out, nullptr, 0);
    k_outinit<<<nblk((long long)n * F2out), TPB, 0, stream>>>(FB, dinv, F32(pbase + 5), FC, n, F2out);
    k_scatter<<<nblk(E * F2out), TPB, 0, stream>>>(ei, ei + E, nullptr, dinv, FB, FC, E, F2out);
    k_relu<<<nblk((long long)n * F2out), TPB, 0, stream>>>(FC, nullptr, n, F2out, F2out);
    // mean pool
    hipMemsetAsync(cnt, 0, (size_t)G * 4, stream);
    hipMemsetAsync(poolOut, 0, (size_t)G * F2out * 4, stream);
    k_count<<<nblk(n), TPB, 0, stream>>>(batch, cnt, n);
    k_pooladd<<<nblk((long long)n * F2out), TPB, 0, stream>>>(FC, batch, poolOut, n, F2out);
    k_pooldiv<<<nblk((long long)G * F2out), TPB, 0, stream>>>(poolOut, cnt, G, F2out);
  };
  molStage(mg_x, ND, N_DRUGS, mg_ei, ED_MOL, mg_batch, p_dgcn, drugS, 78, 312, drugG);
  molStage(pg_x, NT, N_TARGETS, pg_ei, ET_MOL, pg_batch, p_tgcn, targetS, 54, 216, targetG);

  // ---- per-pair branch MLPs ----
  // d1
  k_gather_cvt<<<nblk((long long)N_PAIRS * 256), TPB, 0, stream>>>(drug3, drug_id, ABF,
                                                                   N_PAIRS, 256, 256);
  gemm(ABF, 256, p_ds, N_PAIRS, 1024, 256, true, true, nullptr, 0, ABF2, 1024);
  gemm(ABF2, 1024, p_ds + 2, N_PAIRS, 128, 1024, true, false, nullptr, 0, dpair[0], 128);
  // d2
  k_gather_cvt<<<nblk((long long)N_PAIRS * 320), TPB, 0, stream>>>(drugG, drug_id, ABF,
                                                                   N_PAIRS, 312, 320);
  gemm(ABF, 320, p_dg, N_PAIRS, 1024, 312, true, true, nullptr, 0, ABF2, 1024);
  gemm(ABF2, 1024, p_dg + 2, N_PAIRS, 128, 1024, true, false, nullptr, 0, dpair[1], 128);
  // d3
  k_gather2_cvt<<<nblk((long long)N_PAIRS * 576), TPB, 0, stream>>>(drug3, 256, drugG, 312,
                                                                    drug_id, ABF, N_PAIRS, 576);
  gemm(ABF, 576, p_da, N_PAIRS, 1024, 568, true, true, nullptr, 0, ABF2, 1024);
  gemm(ABF2, 1024, p_da + 2, N_PAIRS, 128, 1024, true, false, nullptr, 0, dpair[2], 128);
  // t1
  k_gather_cvt<<<nblk((long long)N_PAIRS * 256), TPB, 0, stream>>>(target3, target_id, ABF,
                                                                   N_PAIRS, 256, 256);
  gemm(ABF, 256, p_ts, N_PAIRS, 1024, 256, true, true, nullptr, 0, ABF2, 1024);
  gemm(ABF2, 1024, p_ts + 2, N_PAIRS, 128, 1024, true, false, nullptr, 0, tpair[0], 128);
  // t2
  k_gather_cvt<<<nblk((long long)N_PAIRS * 256), TPB, 0, stream>>>(targetG, target_id, ABF,
                                                                   N_PAIRS, 216, 256);
  gemm(ABF, 256, p_tg, N_PAIRS, 1024, 216, true, true, nullptr, 0, ABF2, 1024);
  gemm(ABF2, 1024, p_tg + 2, N_PAIRS, 128, 1024, true, false, nullptr, 0, tpair[1], 128);
  // t3
  k_gather2_cvt<<<nblk((long long)N_PAIRS * 512), TPB, 0, stream>>>(target3, 256, targetG, 216,
                                                                    target_id, ABF, N_PAIRS, 512);
  gemm(ABF, 512, p_ta, N_PAIRS, 1024, 472, true, true, nullptr, 0, ABF2, 1024);
  gemm(ABF2, 1024, p_ta + 2, N_PAIRS, 128, 1024, true, false, nullptr, 0, tpair[2], 128);

  // ---- 9 joint MLPs ----
  for (int i3 = 0; i3 < 3; ++i3) {
    for (int j3 = 0; j3 < 3; ++j3) {
      int jj = 3 * i3 + j3, pb = p_j + 6 * jj;
      k_catbf<<<nblk((long long)N_PAIRS * 256), TPB, 0, stream>>>(dpair[i3], 128, tpair[j3], 128,
                                                                  ABF, N_PAIRS, 256);
      gemm(ABF, 256, pb, N_PAIRS, 1024, 256, true, true, nullptr, 0, ABF2, 1024);
      gemm(ABF2, 1024, pb + 2, N_PAIRS, 512, 1024, true, true, nullptr, 0, ABF3, 512);
      k_dotout<<<nblk(N_PAIRS), TPB, 0, stream>>>(ABF3, 512, F32(pb + 4), F32(pb + 5), ansb,
                                                  N_PAIRS, 512, 9, jj);
    }
  }
  // ---- head ----
  k_final9<<<nblk(N_PAIRS), TPB, 0, stream>>>(ansb, F32(p_mlp), F32(p_mlp + 1), (float*)d_out,
                                              N_PAIRS);
}
